// DRPGAT_89412629168235
// MI455X (gfx1250) — matrix operand order repaired and checked by reading, not compiled, not run on hardware
//
#include <hip/hip_runtime.h>
#include <hip/hip_bf16.h>

typedef __attribute__((ext_vector_type(16))) __bf16 v16bf;
typedef __attribute__((ext_vector_type(4)))  __bf16 v4bf;
typedef __attribute__((ext_vector_type(8)))  float  v8f;

#define N_NODE  4096
#define IN_DIM  200
#define OUT_DIM 200
#define N_HEADS 8
#define SEQ_LEN 4
#define HID     200

#define ACT_NONE    0
#define ACT_ELU     1
#define ACT_SIGMOID 2
#define ACT_TANH    3

// ---------------------------------------------------------------------------
// Staging register block: A tile 128x32 (4 float4/thread), B tile 32x64
// (2 float4/thread), 256 threads.
// ---------------------------------------------------------------------------
__device__ __forceinline__ void load_stage_regs(
    const float* __restrict__ Ab, const float* __restrict__ B,
    int M, int N, int K, int lda, int ldb,
    int tileM, int tileN, int k0, int tid,
    float4* ar, float4* br)
{
    const bool fastA = (tileM + 128 <= M) && (k0 + 32 <= K);
    const bool fastB = (tileN + 64 <= N) && (k0 + 32 <= K);

#pragma unroll
    for (int p = 0; p < 4; ++p) {
        int idx = tid + p * 256;
        int r   = idx >> 3;          // 8 float4 per 32-wide row
        int c   = (idx & 7) * 4;
        if (fastA) {
            ar[p] = *(const float4*)(Ab + (long long)(tileM + r) * lda + k0 + c);
        } else {
            float4 v = {0.f, 0.f, 0.f, 0.f};
            int gr = tileM + r;
            if (gr < M) {
                const float* src = Ab + (long long)gr * lda + k0 + c;
                if (k0 + c + 0 < K) v.x = src[0];
                if (k0 + c + 1 < K) v.y = src[1];
                if (k0 + c + 2 < K) v.z = src[2];
                if (k0 + c + 3 < K) v.w = src[3];
            }
            ar[p] = v;
        }
    }
#pragma unroll
    for (int p = 0; p < 2; ++p) {
        int idx = tid + p * 256;
        int kr  = idx >> 4;          // 16 float4 per 64-wide row
        int c   = (idx & 15) * 4;
        if (fastB) {
            br[p] = *(const float4*)(B + (long long)(k0 + kr) * ldb + tileN + c);
        } else {
            float4 v = {0.f, 0.f, 0.f, 0.f};
            if (k0 + kr < K) {
                const float* src = B + (long long)(k0 + kr) * ldb + tileN + c;
                if (tileN + c + 0 < N) v.x = src[0];
                if (tileN + c + 1 < N) v.y = src[1];
                if (tileN + c + 2 < N) v.z = src[2];
                if (tileN + c + 3 < N) v.w = src[3];
            }
            br[p] = v;
        }
    }
}

// ---------------------------------------------------------------------------
// General GEMM: C[b] = act( A[b] @ B + bias_row + addmat )
//   A: [M,K] f32 row-major (lda), batch stride strideA
//   B: [K,N] f32 row-major (ldb), shared across batch
//   C: [M,N] f32 row-major (ldc), batch stride strideC
// Block tile 128x64, 256 threads = 8 waves (4x2), each wave 32x32 via
// 2x2 v_wmma_f32_16x16x32_bf16; f32->bf16 conversion in the staging path;
// K loop software-pipelined (global loads for slice i+1 overlap WMMA of i).
// ---------------------------------------------------------------------------
__global__ __launch_bounds__(256)
void gemm_bf16_wmma(const float* __restrict__ A, const float* __restrict__ B,
                    float* __restrict__ C,
                    int M, int N, int K,
                    int lda, int ldb, int ldc,
                    long long strideA, long long strideC,
                    const float* __restrict__ bias,
                    const float* __restrict__ addmat, int addld,
                    int act)
{
    __shared__ __bf16 As[128][32];   // [m][k]
    __shared__ __bf16 Bs[64][32];    // transposed: [n][k]

    const int tid  = threadIdx.x;
    const int lane = tid & 31;
    const int wave = tid >> 5;
    const int half = lane >> 4;      // 0 or 1
    const int l16  = lane & 15;
    const int wm   = wave >> 1;      // 0..3  (row group of 32)
    const int wn   = wave & 1;       // 0..1  (col group of 32)

    const int tileM = blockIdx.x * 128;
    const int tileN = blockIdx.y * 64;
    const int batch = blockIdx.z;

    const float* Ab = A + (long long)batch * strideA;
    float*       Cb = C + (long long)batch * strideC;

    v8f acc[2][2];
    v8f zero = {0.f, 0.f, 0.f, 0.f, 0.f, 0.f, 0.f, 0.f};
#pragma unroll
    for (int i = 0; i < 2; ++i)
#pragma unroll
        for (int j = 0; j < 2; ++j) acc[i][j] = zero;

    float4 ar[4], br[2];
    load_stage_regs(Ab, B, M, N, K, lda, ldb, tileM, tileN, 0, tid, ar, br);

    for (int k0 = 0; k0 < K; k0 += 32) {
        if (k0) __syncthreads();     // prev fragment reads complete before overwrite

        // ---- store staged regs -> LDS (bf16) ----
#pragma unroll
        for (int p = 0; p < 4; ++p) {
            int idx = tid + p * 256;
            int r   = idx >> 3;
            int c   = (idx & 7) * 4;
            v4bf pk = { (__bf16)ar[p].x, (__bf16)ar[p].y,
                        (__bf16)ar[p].z, (__bf16)ar[p].w };
            *(v4bf*)&As[r][c] = pk;
        }
#pragma unroll
        for (int p = 0; p < 2; ++p) {
            int idx = tid + p * 256;
            int kr  = idx >> 4;
            int c   = (idx & 15) * 4;
            Bs[c + 0][kr] = (__bf16)br[p].x;
            Bs[c + 1][kr] = (__bf16)br[p].y;
            Bs[c + 2][kr] = (__bf16)br[p].z;
            Bs[c + 3][kr] = (__bf16)br[p].w;
        }
        __syncthreads();

        // ---- prefetch next K-slice into regs (overlaps WMMA below) ----
        if (k0 + 32 < K)
            load_stage_regs(Ab, B, M, N, K, lda, ldb, tileM, tileN, k0 + 32,
                            tid, ar, br);

        // ---- per-lane WMMA fragments ----
        // A 16x32 bf16: row=l16; lanes 0-15 K=0..7,16..23; lanes 16-31 K=8..15,24..31
        v16bf afrag[2], bfrag[2];
#pragma unroll
        for (int mi = 0; mi < 2; ++mi) {
            int row = wm * 32 + mi * 16 + l16;
            int c0  = 8 * half;
#pragma unroll
            for (int e = 0; e < 8; ++e) afrag[mi][e]     = As[row][c0 + e];
#pragma unroll
            for (int e = 0; e < 8; ++e) afrag[mi][8 + e] = As[row][16 + c0 + e];
        }
        // B 32x16 bf16: col=l16; lanes 0-15 K=0..7,16..23; lanes 16-31 K=8..15,24..31
#pragma unroll
        for (int ni = 0; ni < 2; ++ni) {
            int col = wn * 32 + ni * 16 + l16;
#pragma unroll
            for (int e = 0; e < 8; ++e) bfrag[ni][e] = Bs[col][8 * half + e];
#pragma unroll
            for (int e = 0; e < 8; ++e) bfrag[ni][8 + e] = Bs[col][16 + 8 * half + e];
        }

#pragma unroll
        for (int mi = 0; mi < 2; ++mi)
#pragma unroll
            for (int ni = 0; ni < 2; ++ni)
                acc[mi][ni] = __builtin_amdgcn_wmma_f32_16x16x32_bf16(
                    false, afrag[mi], false, bfrag[ni],
                    (short)0, acc[mi][ni], false, false);
    }

    // ---- epilogue: bias / addmat / activation / store ----
#pragma unroll
    for (int mi = 0; mi < 2; ++mi) {
#pragma unroll
        for (int ni = 0; ni < 2; ++ni) {
            int col = tileN + wn * 32 + ni * 16 + l16;
#pragma unroll
            for (int v = 0; v < 8; ++v) {
                int row = tileM + wm * 32 + mi * 16 + v + 8 * half;
                if (row < M && col < N) {
                    float x = acc[mi][ni][v];
                    if (bias)   x += bias[col];
                    if (addmat) x += addmat[(long long)row * addld + col];
                    if (act == ACT_ELU)          x = (x > 0.0f) ? x : (__expf(x) - 1.0f);
                    else if (act == ACT_SIGMOID) x = 1.0f / (1.0f + __expf(-x));
                    else if (act == ACT_TANH)    x = tanhf(x);
                    Cb[(long long)row * ldc + col] = x;
                }
            }
        }
    }
}

// ---------------------------------------------------------------------------
// logits[n,o] = outflat[o*N_NODE + n]  (reshape(200,4096).T reinterpretation)
// softmax over o; one block (256 thr) per node row n.
// ---------------------------------------------------------------------------
__global__ __launch_bounds__(256)
void softmax_reinterp(const float* __restrict__ outflat, float* __restrict__ dst)
{
    __shared__ float red[256];
    int n = blockIdx.x;
    int o = threadIdx.x;
    float x = (o < OUT_DIM) ? outflat[(long long)o * N_NODE + n] : -3.4e38f;
    red[o] = x; __syncthreads();
    for (int s = 128; s > 0; s >>= 1) {
        if (o < s) red[o] = fmaxf(red[o], red[o + s]);
        __syncthreads();
    }
    float m = red[0]; __syncthreads();
    float e = (o < OUT_DIM) ? __expf(x - m) : 0.0f;
    red[o] = e; __syncthreads();
    for (int s = 128; s > 0; s >>= 1) {
        if (o < s) red[o] += red[o + s];
        __syncthreads();
    }
    float sum = red[0];
    if (o < OUT_DIM) dst[(long long)n * OUT_DIM + o] = e / sum;
}

// W1t[k][o] = W1[o][k];  W2sT[k][o] = sum_h W2[o, h*HID + k]
__global__ void prep_small(const float* __restrict__ W1, const float* __restrict__ W2,
                           float* __restrict__ W1t, float* __restrict__ W2sT)
{
    int idx = blockIdx.x * 256 + threadIdx.x;
    if (idx >= IN_DIM * HID) return;
    int k = idx / HID, o = idx % HID;
    W1t[k * HID + o] = W1[o * IN_DIM + k];
    float s = 0.0f;
#pragma unroll
    for (int h = 0; h < N_HEADS; ++h) s += W2[o * (N_HEADS * HID) + h * HID + k];
    W2sT[k * OUT_DIM + o] = s;
}

// Wcat[n][0:200]=Wr, [200:400]=Wu, [400:600]=Wh
__global__ void prep_wcat(const float* __restrict__ Wr, const float* __restrict__ Wu,
                          const float* __restrict__ Wh, float* __restrict__ Wcat)
{
    int idx = blockIdx.x * 256 + threadIdx.x;
    if (idx >= N_NODE * OUT_DIM) return;
    int n = idx / OUT_DIM, o = idx % OUT_DIM;
    long long base = (long long)n * 600;
    Wcat[base + o]       = Wr[idx];
    Wcat[base + 200 + o] = Wu[idx];
    Wcat[base + 400 + o] = Wh[idx];
}

__global__ void ew_mul(const float* __restrict__ a, const float* __restrict__ b,
                       float* __restrict__ c, int n)
{
    int i = blockIdx.x * 256 + threadIdx.x;
    if (i < n) c[i] = a[i] * b[i];
}

// wv = (1-u)*wv + u*hcap
__global__ void ew_gru(const float* __restrict__ u, const float* __restrict__ hcap,
                       float* __restrict__ wv, int n)
{
    int i = blockIdx.x * 256 + threadIdx.x;
    if (i < n) {
        float ui = u[i];
        wv[i] = (1.0f - ui) * wv[i] + ui * hcap[i];
    }
}

// ---------------------------------------------------------------------------
extern "C" void kernel_launch(void* const* d_in, const int* in_sizes, int n_in,
                              void* d_out, int out_size, void* d_ws, size_t ws_size,
                              hipStream_t stream)
{
    (void)in_sizes; (void)n_in; (void)out_size; (void)ws_size;

    const float* adjs    = (const float*)d_in[0];   // [4,4096,4096]
    const float* feats   = (const float*)d_in[1];   // [4,4096,200]
    const float* init_wv = (const float*)d_in[3];   // [8,200,200]
    const float* W1      = (const float*)d_in[4];
    const float* b1      = (const float*)d_in[5];
    const float* W2      = (const float*)d_in[10];
    const float* b2      = (const float*)d_in[11];
    const float* Wr      = (const float*)d_in[16];
    const float* Ur      = (const float*)d_in[17];
    const float* Pr      = (const float*)d_in[18];
    const float* br      = (const float*)d_in[19];
    const float* Wu      = (const float*)d_in[20];
    const float* Uu      = (const float*)d_in[21];
    const float* Pu      = (const float*)d_in[22];
    const float* bu      = (const float*)d_in[23];
    const float* Wh      = (const float*)d_in[24];
    const float* Uh      = (const float*)d_in[25];
    const float* Ph      = (const float*)d_in[26];
    const float* bh      = (const float*)d_in[27];

    // workspace layout (f32 elements; all offsets multiples of 4 -> 16B aligned)
    float* ws = (float*)d_ws;
    size_t off = 0;
    float* W1t  = ws + off; off += 200 * 200;
    float* W2sT = ws + off; off += 200 * 200;
    float* Wcat = ws + off; off += (size_t)N_NODE * 600;
    float* Hbuf = ws + off; off += (size_t)N_NODE * HID;
    float* Obuf = ws + off; off += (size_t)N_NODE * OUT_DIM;
    float* AW   = ws + off; off += (size_t)N_NODE * 600;
    float* G    = ws + off; off += 3 * 200 * 200;     // gr|gu|gh
    float* wv   = ws + off; off += N_HEADS * 200 * 200;
    float* Rb   = ws + off; off += N_HEADS * 200 * 200;
    float* Ub   = ws + off; off += N_HEADS * 200 * 200;
    float* RW   = ws + off; off += N_HEADS * 200 * 200;
    float* Hc   = ws + off; off += N_HEADS * 200 * 200;

    float* embeds  = (float*)d_out;                                   // [4,4096,200]
    float* finalwv = embeds + (size_t)SEQ_LEN * N_NODE * OUT_DIM;     // [8,200,200]

    const int WVN = N_HEADS * 200 * 200;   // 320000

    // ---- prep ----
    prep_small<<<(200 * 200 + 255) / 256, 256, 0, stream>>>(W1, W2, W1t, W2sT);
    prep_wcat<<<(N_NODE * OUT_DIM + 255) / 256, 256, 0, stream>>>(Wr, Wu, Wh, Wcat);
    hipMemcpyAsync(wv, init_wv, (size_t)WVN * sizeof(float),
                   hipMemcpyDeviceToDevice, stream);

    // ---- GAT embeds, per timestep ----
    dim3 gGAT((N_NODE + 127) / 128, (OUT_DIM + 63) / 64, 1);
    for (int t = 0; t < SEQ_LEN; ++t) {
        const float* xt = feats + (size_t)t * N_NODE * IN_DIM;
        gemm_bf16_wmma<<<gGAT, 256, 0, stream>>>(
            xt, W1t, Hbuf, N_NODE, HID, IN_DIM, IN_DIM, HID, HID,
            0LL, 0LL, b1, nullptr, 0, ACT_ELU);
        gemm_bf16_wmma<<<gGAT, 256, 0, stream>>>(
            Hbuf, W2sT, Obuf, N_NODE, OUT_DIM, HID, HID, OUT_DIM, OUT_DIM,
            0LL, 0LL, b2, nullptr, 0, ACT_ELU);
        softmax_reinterp<<<N_NODE, 256, 0, stream>>>(
            Obuf, embeds + (size_t)t * N_NODE * OUT_DIM);
    }

    // ---- GRU over t = 1..3 (sequential) ----
    dim3 gAW((N_NODE + 127) / 128, (600 + 63) / 64, 1);
    dim3 gGm((200 + 127) / 128, (200 + 63) / 64, 1);
    dim3 gBh((200 + 127) / 128, (200 + 63) / 64, N_HEADS);
    const float* Ps[3] = {Pr, Pu, Ph};
    const float* bs[3] = {br, bu, bh};

    for (int s = 0; s < 3; ++s) {
        const float* adj = adjs + (size_t)(s + 1) * N_NODE * N_NODE;
        // AW = adj @ [Wr|Wu|Wh]   (dominant GEMM: M=4096,N=600,K=4096)
        gemm_bf16_wmma<<<gAW, 256, 0, stream>>>(
            adj, Wcat, AW, N_NODE, 600, N_NODE, N_NODE, 600, 600,
            0LL, 0LL, nullptr, nullptr, 0, ACT_NONE);
        // Gg = Pg @ AW[:, g*200:(g+1)*200] + bg
        for (int g = 0; g < 3; ++g) {
            gemm_bf16_wmma<<<gGm, 256, 0, stream>>>(
                Ps[g], AW + g * 200, G + g * 200 * 200,
                200, 200, N_NODE, N_NODE, 600, 200,
                0LL, 0LL, nullptr, bs[g], 200, ACT_NONE);
        }
        // r = sigmoid(Gr + wv @ Ur)    (batched over 8 heads)
        gemm_bf16_wmma<<<gBh, 256, 0, stream>>>(
            wv, Ur, Rb, 200, 200, 200, 200, 200, 200,
            40000LL, 40000LL, nullptr, G, 200, ACT_SIGMOID);
        // u = sigmoid(Gu + wv @ Uu)
        gemm_bf16_wmma<<<gBh, 256, 0, stream>>>(
            wv, Uu, Ub, 200, 200, 200, 200, 200, 200,
            40000LL, 40000LL, nullptr, G + 40000, 200, ACT_SIGMOID);
        // RW = r * wv
        ew_mul<<<(WVN + 255) / 256, 256, 0, stream>>>(Rb, wv, RW, WVN);
        // hcap = tanh(Gh + RW @ Uh)
        gemm_bf16_wmma<<<gBh, 256, 0, stream>>>(
            RW, Uh, Hc, 200, 200, 200, 200, 200, 200,
            40000LL, 40000LL, nullptr, G + 80000, 200, ACT_TANH);
        // wv = (1-u)*wv + u*hcap
        ew_gru<<<(WVN + 255) / 256, 256, 0, stream>>>(Ub, Hc, wv, WVN);
    }

    hipMemcpyAsync(finalwv, wv, (size_t)WVN * sizeof(float),
                   hipMemcpyDeviceToDevice, stream);
}
